// GruRnn_75333726372314
// MI455X (gfx1250) — compile-verified, matrix operand order checked by reading, not run on hardware
//
#include <hip/hip_runtime.h>

typedef __bf16 bf16;
typedef bf16  v16bf __attribute__((ext_vector_type(16)));
typedef bf16  v8bf  __attribute__((ext_vector_type(8)));
typedef float v8f   __attribute__((ext_vector_type(8)));

#define GRU_B   64
#define GRU_T   1024
#define GRU_DIN 256
#define GRU_H   512
#define GRU_G   1536   // 3*H

__device__ __forceinline__ bf16 f2bf(float f) { return (bf16)f; }

// ---------------------------------------------------------------------------
// Fragment loaders per CDNA5 wave32 WMMA layouts (CDNA5 ISA §7.12.2).
// 16-bit A (16x32): lane = m + 16*half; VGPR0-3 hold K = 8*half + 0..7,
// VGPR4-7 hold K = 16 + 8*half + 0..7  -> two contiguous 8-element runs.
// B (32x16) uses the symmetric layout with lane = n + 16*half; since our
// weight matrices are row-major [N,K], B-fragment rows are contiguous in K.
// ---------------------------------------------------------------------------
__device__ __forceinline__ v16bf load_frag_bf16(const bf16* __restrict__ base,
                                                int row, int ldk, int k0, int half) {
  union { v16bf v; v8bf h[2]; } u;
  const bf16* p = base + (size_t)row * ldk + k0 + (half << 3);
  u.h[0] = *(const v8bf*)(p);        // K = k0 + 8*half + 0..7
  u.h[1] = *(const v8bf*)(p + 16);   // K = k0 + 16 + 8*half + 0..7
  return u.v;
}

__device__ __forceinline__ v16bf load_frag_f32(const float* __restrict__ base,
                                               int row, int ldk, int k0, int half) {
  v16bf a;
  const float* p = base + (size_t)row * ldk + k0 + (half << 3);
#pragma unroll
  for (int e = 0; e < 8; ++e)  a[e]     = f2bf(p[e]);
#pragma unroll
  for (int e = 0; e < 8; ++e)  a[8 + e] = f2bf(p[16 + e]);
  return a;
}

// ---------------------------------------------------------------------------
// fp32 -> bf16 elementwise convert
// ---------------------------------------------------------------------------
__global__ void cvt_bf16_kernel(const float* __restrict__ src, bf16* __restrict__ dst, int n) {
  int i = blockIdx.x * blockDim.x + threadIdx.x;
  int stride = gridDim.x * blockDim.x;
  for (; i < n; i += stride) dst[i] = f2bf(src[i]);
}

// ---------------------------------------------------------------------------
// Input projection: xg[R,G] = X[R,K] * W[G,K]^T + bias[G]   (fp32 out)
// One wave computes a 16x64 tile (A fragment reused across 4 N-tiles).
// ---------------------------------------------------------------------------
template <bool SRC_BF>
__global__ void input_proj_kernel(const void* __restrict__ xsrc,
                                  const bf16* __restrict__ wbf,   // [G,K] bf16 row-major
                                  const float* __restrict__ bias, // [G]
                                  float* __restrict__ xg,         // [R,G]
                                  int R, int K, int G) {
  const int lane = threadIdx.x & 31;
  const int wave = blockIdx.x * (blockDim.x >> 5) + (threadIdx.x >> 5);
  const int ntGroups = G >> 6;                  // N-tile groups of 4
  const int mt = wave / ntGroups;
  const int ng = wave % ntGroups;
  if (mt >= (R >> 4)) return;                   // wave-uniform
  const int row0 = mt << 4;
  const int col0 = ng << 6;
  const int half = lane >> 4;
  const int mr   = lane & 15;

  const float* xf = (const float*)xsrc;
  const bf16*  xb = (const bf16*)xsrc;

  v8f acc[4] = {};
  for (int k0 = 0; k0 < K; k0 += 32) {
    v16bf a = SRC_BF ? load_frag_bf16(xb, row0 + mr, K, k0, half)
                     : load_frag_f32 (xf, row0 + mr, K, k0, half);
#pragma unroll
    for (int j = 0; j < 4; ++j) {
      v16bf b = load_frag_bf16(wbf, col0 + (j << 4) + mr, K, k0, half);
      acc[j] = __builtin_amdgcn_wmma_f32_16x16x32_bf16(
          false, a, false, b, (short)0, acc[j], false, false);
    }
  }
  // D layout: VGPR r -> M = r + 8*half, N = lane&15
#pragma unroll
  for (int j = 0; j < 4; ++j) {
    int col = col0 + (j << 4) + mr;
    float bs = bias[col];
#pragma unroll
    for (int r = 0; r < 8; ++r) {
      int m = row0 + r + (half << 3);
      xg[(size_t)m * G + col] = acc[j][r] + bs;
    }
  }
}

// ---------------------------------------------------------------------------
// Grid-wide barrier (monotonic counter, agent scope)
// ---------------------------------------------------------------------------
__device__ __forceinline__ void grid_sync(unsigned* ctr, unsigned target) {
  __syncthreads();
  if (threadIdx.x == 0) {
    __threadfence();
    __hip_atomic_fetch_add(ctr, 1u, __ATOMIC_RELEASE, __HIP_MEMORY_SCOPE_AGENT);
    while (__hip_atomic_load(ctr, __ATOMIC_ACQUIRE, __HIP_MEMORY_SCOPE_AGENT) < target)
      __builtin_amdgcn_s_sleep(2);
  }
  __syncthreads();
  __threadfence();
}

// ---------------------------------------------------------------------------
// Persistent recurrent scan for one layer.
// Grid MUST be 48 blocks x 256 threads = 384 waves = 4 M-tiles x 96 N-tiles.
// Per step: hg = h_bf(64x512) x W_hh^T -> gates -> h update.
// ---------------------------------------------------------------------------
__global__ void gru_scan_kernel(const float* __restrict__ xg,   // [B*T, G], row r = b*T + t
                                const bf16*  __restrict__ whh,  // [G, H] bf16
                                const float* __restrict__ bhh,  // [G]
                                float* __restrict__ hg,         // [B, G] scratch
                                float* __restrict__ hst,        // [B, H] fp32 state
                                bf16*  __restrict__ hbf,        // [B, H] bf16 state
                                bf16*  __restrict__ ybf,        // [B*T, H] or null
                                float* __restrict__ yf,         // [B*T, H] or null
                                float* __restrict__ hlast,      // [B, H]
                                unsigned* __restrict__ ctr,
                                int T) {
  const int lane = threadIdx.x & 31;
  const int wave = blockIdx.x * (blockDim.x >> 5) + (threadIdx.x >> 5); // 0..383
  const int mt = wave / 96;          // 0..3   (batch tile)
  const int nt = wave % 96;          // 0..95  (gate tile)
  const int half = lane >> 4;
  const int mr   = lane & 15;
  const int row0 = mt << 4;
  const int col0 = nt << 4;
  const int tid  = blockIdx.x * blockDim.x + threadIdx.x;
  const int nthreads = gridDim.x * blockDim.x;
  const unsigned nblocks = gridDim.x;

  // h0 = 0
  for (int i = tid; i < GRU_B * GRU_H; i += nthreads) {
    hst[i] = 0.0f;
    hbf[i] = (bf16)0.0f;
  }
  unsigned bar = 0;
  bar += 1; grid_sync(ctr, bar * nblocks);

  for (int t = 0; t < T; ++t) {
    // ---- recurrent GEMM: 16 chained WMMAs over K=512 ----
    v8f acc = {};
#pragma unroll 4
    for (int k0 = 0; k0 < GRU_H; k0 += 32) {
      v16bf a = load_frag_bf16(hbf, row0 + mr, GRU_H, k0, half);
      v16bf b = load_frag_bf16(whh, col0 + mr, GRU_H, k0, half);
      acc = __builtin_amdgcn_wmma_f32_16x16x32_bf16(
          false, a, false, b, (short)0, acc, false, false);
    }
#pragma unroll
    for (int r = 0; r < 8; ++r)
      hg[(size_t)(row0 + r + (half << 3)) * GRU_G + col0 + mr] = acc[r];

    bar += 1; grid_sync(ctr, bar * nblocks);

    // ---- gates + state update (fp32) ----
    for (int i = tid; i < GRU_B * GRU_H; i += nthreads) {
      int b_ = i >> 9;          // /512
      int h_ = i & 511;
      const float* xrow = xg + (size_t)(b_ * T + t) * GRU_G;
      float xr = xrow[h_];
      float xz = xrow[GRU_H + h_];
      float xn = xrow[2 * GRU_H + h_];
      const float* hrow = hg + (size_t)b_ * GRU_G;
      float hr = hrow[h_]             + bhh[h_];
      float hz = hrow[GRU_H + h_]     + bhh[GRU_H + h_];
      float hn = hrow[2 * GRU_H + h_] + bhh[2 * GRU_H + h_];
      float r = 1.0f / (1.0f + __expf(-(xr + hr)));
      float z = 1.0f / (1.0f + __expf(-(xz + hz)));
      float n = tanhf(xn + r * hn);
      float hnew = (1.0f - z) * n + z * hst[i];
      hst[i] = hnew;
      hbf[i] = f2bf(hnew);
      size_t yi = (size_t)(b_ * T + t) * GRU_H + h_;
      if (yf)  yf[yi]  = hnew;
      if (ybf) ybf[yi] = f2bf(hnew);
      if (t == T - 1) hlast[i] = hnew;
    }
    bar += 1; grid_sync(ctr, bar * nblocks);
  }
}

// ---------------------------------------------------------------------------
extern "C" void kernel_launch(void* const* d_in, const int* in_sizes, int n_in,
                              void* d_out, int out_size, void* d_ws, size_t ws_size,
                              hipStream_t stream) {
  (void)in_sizes; (void)n_in; (void)out_size; (void)ws_size;
  const int B = GRU_B, T = GRU_T, DIN = GRU_DIN, H = GRU_H, G = GRU_G, L = 3;
  const int R = B * T;

  const float* x_in = (const float*)d_in[0];
  const float* w_ih[3] = {(const float*)d_in[1], (const float*)d_in[5], (const float*)d_in[9]};
  const float* w_hh[3] = {(const float*)d_in[2], (const float*)d_in[6], (const float*)d_in[10]};
  const float* b_ih[3] = {(const float*)d_in[3], (const float*)d_in[7], (const float*)d_in[11]};
  const float* b_hh[3] = {(const float*)d_in[4], (const float*)d_in[8], (const float*)d_in[12]};

  // workspace carve-up (256B aligned)
  char* ws = (char*)d_ws;
  size_t off = 0;
  auto alloc = [&](size_t bytes) -> void* {
    void* p = ws + off;
    off = (off + bytes + 255) & ~(size_t)255;
    return p;
  };
  float* xg   = (float*)alloc((size_t)R * G * sizeof(float));     // 403 MB, reused/layer
  bf16*  y0   = (bf16*) alloc((size_t)R * H * sizeof(bf16));      // layer0 -> layer1 acts
  bf16*  y1   = (bf16*) alloc((size_t)R * H * sizeof(bf16));      // layer1 -> layer2 acts
  float* hg   = (float*)alloc((size_t)B * G * sizeof(float));
  float* hst  = (float*)alloc((size_t)B * H * sizeof(float));
  bf16*  hbf  = (bf16*) alloc((size_t)B * H * sizeof(bf16));
  bf16*  wihb = (bf16*) alloc((size_t)G * H * sizeof(bf16));      // max K = H
  bf16*  whhb = (bf16*) alloc((size_t)G * H * sizeof(bf16));
  unsigned* ctr = (unsigned*)alloc(256);

  float* y_final = (float*)d_out;                      // [B,T,H]
  float* h_stack = (float*)d_out + (size_t)B * T * H;  // [L,B,H]

  for (int l = 0; l < L; ++l) {
    const int K = (l == 0) ? DIN : H;

    // weights -> bf16 (W_hh stays L2-resident across the whole scan)
    cvt_bf16_kernel<<<384, 256, 0, stream>>>(w_ih[l], wihb, G * K);
    cvt_bf16_kernel<<<384, 256, 0, stream>>>(w_hh[l], whhb, G * H);

    // hoisted input projection: xg = X * W_ih^T + b_ih
    const int totalWaves = (R / 16) * (G / 64);   // 98304
    const int blocks = totalWaves / 8;            // 12288 blocks of 8 waves
    if (l == 0) {
      input_proj_kernel<false><<<blocks, 256, 0, stream>>>(
          (const void*)x_in, wihb, b_ih[l], xg, R, K, G);
    } else {
      input_proj_kernel<true><<<blocks, 256, 0, stream>>>(
          (const void*)(l == 1 ? y0 : y1), wihb, b_ih[l], xg, R, K, G);
    }

    // persistent sequential scan over T
    hipMemsetAsync(ctr, 0, sizeof(unsigned), stream);
    bf16*  yb  = (l == 0) ? y0 : (l == 1 ? y1 : nullptr);
    float* yfp = (l == 2) ? y_final : nullptr;
    gru_scan_kernel<<<48, 256, 0, stream>>>(
        xg, whhb, b_hh[l], hg, hst, hbf, yb, yfp,
        h_stack + (size_t)l * B * H, ctr, T);
  }
}
